// ContrastiveLearningLoss_81406810128730
// MI455X (gfx1250) — matrix operand order repaired and checked by reading, not compiled, not run on hardware
//
#include <hip/hip_runtime.h>
#include <hip/hip_bf16.h>
#include <math.h>

// ---------------------------------------------------------------------------
// Problem constants (from the reference)
//   B=16, L=512, DIN1=256, DIN2=192, DH=256, DF=128, N=B*L=8192
// ---------------------------------------------------------------------------

typedef __attribute__((ext_vector_type(16))) __bf16 v16bf;
typedef __attribute__((ext_vector_type(8)))  float  v8f;

#define LOG2E_F 1.4426950408889634f
#define LN2_F   0.6931471805599453f
#define CSPLIT  8          // column-chunk parallelism for the big GEMM+lse

__device__ __forceinline__ unsigned short f32_to_bf16(float x) {
  unsigned int u = __float_as_uint(x);
  u += 0x7FFFu + ((u >> 16) & 1u);      // round-to-nearest-even
  return (unsigned short)(u >> 16);
}

union Frag16 { uint4 u[2]; v16bf v; };

// A-fragment for V_WMMA_*_16X16X32_BF16 (16x32, row-major source).
// lane: m = lane&15 ; h = lane>=16
// elems 0..7  -> K = kbase + 8h + 0..7
// elems 8..15 -> K = kbase + 16 + 8h + 0..7
__device__ __forceinline__ v16bf load_a_frag(const unsigned short* __restrict__ A,
                                             int row, int K, int kbase, int h) {
  const unsigned short* base = A + (size_t)row * K + kbase;
  Frag16 f;
  f.u[0] = *(const uint4*)(base + 8 * h);
  f.u[1] = *(const uint4*)(base + 16 + 8 * h);
  return f.v;
}

// B-fragment (32x16) loaded from Bt = B transposed (N x K row-major).
// lane: n = lane&15 ; elems 0..7 -> K = kbase + 8h + 0..7, elems 8..15 -> K = kbase + 16 + 8h + 0..7
__device__ __forceinline__ v16bf load_b_frag(const unsigned short* __restrict__ Bt,
                                             int col, int K, int kbase, int h) {
  const unsigned short* base = Bt + (size_t)col * K + kbase;
  Frag16 f;
  f.u[0] = *(const uint4*)(base + 8 * h);
  f.u[1] = *(const uint4*)(base + 16 + 8 * h);
  return f.v;
}

// ---------------------------------------------------------------------------
// Elementwise f32 -> bf16
// ---------------------------------------------------------------------------
__global__ void cvt_bf16_kernel(const float* __restrict__ src,
                                unsigned short* __restrict__ dst, int n) {
  int i = blockIdx.x * blockDim.x + threadIdx.x;
  if (i < n) dst[i] = f32_to_bf16(src[i]);
}

// Transpose RxC f32 -> CxR bf16 (weights are tiny; simple is fine)
__global__ void transpose_bf16_kernel(const float* __restrict__ src,
                                      unsigned short* __restrict__ dst, int R, int C) {
  int i = blockIdx.x * blockDim.x + threadIdx.x;
  if (i < R * C) {
    int r = i / C, c = i % C;
    dst[(size_t)c * R + r] = f32_to_bf16(src[i]);
  }
}

// ---------------------------------------------------------------------------
// Generic bf16 WMMA GEMM:  out = [relu](A @ Bt^T + bias)
//   A  : N x K  bf16 row-major
//   Bt : M x K  bf16 row-major  (i.e. B transposed)
//   block = 128 threads (4 waves), wave -> one 16x16 output tile
//   grid  = (M/64, Nrows/16)
// ---------------------------------------------------------------------------
__global__ void gemm_bf16_kernel(const unsigned short* __restrict__ A,
                                 const unsigned short* __restrict__ Bt,
                                 const float* __restrict__ bias,
                                 float* __restrict__ outF,
                                 unsigned short* __restrict__ outB,
                                 int K, int M, int relu) {
  const int lane = threadIdx.x & 31;
  const int wave = threadIdx.x >> 5;
  const int h = lane >> 4;
  const int ln = lane & 15;
  const int rowbase = blockIdx.y * 16;
  const int colbase = (blockIdx.x * 4 + wave) * 16;

  const int row = rowbase + ln;
  const int col = colbase + ln;

  v8f acc = {};
  for (int kb = 0; kb < K; kb += 32) {
    v16bf a = load_a_frag(A, row, K, kb, h);
    v16bf b = load_b_frag(Bt, col, K, kb, h);
    acc = __builtin_amdgcn_wmma_f32_16x16x32_bf16(false, a, false, b,
                                                  (short)0, acc, false, false);
  }

  const float bv = bias[col];
#pragma unroll
  for (int r = 0; r < 8; ++r) {
    int m = rowbase + r + 8 * h;
    float v = acc[r] + bv;
    if (relu) v = fmaxf(v, 0.0f);
    size_t idx = (size_t)m * M + col;
    if (outF) outF[idx] = v;
    if (outB) outB[idx] = f32_to_bf16(v);
  }
}

// ---------------------------------------------------------------------------
// positive[i] = dot(e1[i],e2[i]) + band_mean(e1·e1ᵀ) + band_mean(e2·e2ᵀ)
//                                + band_mean(e1·e2ᵀ)     (per-batch bands)
// One wave per row; lane owns 4 of the 128 features.
// ---------------------------------------------------------------------------
__global__ void band_positive_kernel(const float* __restrict__ e1,
                                     const float* __restrict__ e2,
                                     float* __restrict__ pos,
                                     const int* __restrict__ pr_self,
                                     const int* __restrict__ pr_tgt,
                                     int L) {
  const int lane = threadIdx.x & 31;
  const int wave = threadIdx.x >> 5;
  const int row = blockIdx.x * 8 + wave;
  const int b = row / L, j = row % L;
  const int rs = *pr_self, rt = *pr_tgt;

  const float4 a1 = ((const float4*)(e1 + (size_t)row * 128))[lane];
  const float4 a2 = ((const float4*)(e2 + (size_t)row * 128))[lane];
  float total = a1.x * a2.x + a1.y * a2.y + a1.z * a2.z + a1.w * a2.w;

  if (rs > 0) {
    int lo = max(j - rs, 0), hi = min(j + rs + 1, L);
    float s = 0.f;
    for (int m = lo; m < hi; ++m) {
      size_t r2 = ((size_t)b * L + m) * 128;
      float4 x1 = ((const float4*)(e1 + r2))[lane];
      float4 x2 = ((const float4*)(e2 + r2))[lane];
      s += a1.x * x1.x + a1.y * x1.y + a1.z * x1.z + a1.w * x1.w;
      s += a2.x * x2.x + a2.y * x2.y + a2.z * x2.z + a2.w * x2.w;
    }
    total += s / (float)(hi - lo);
  }
  if (rt > 0) {
    int lo = max(j - rt, 0), hi = min(j + rt + 1, L);
    float s = 0.f;
    for (int m = lo; m < hi; ++m) {
      size_t r2 = ((size_t)b * L + m) * 128;
      float4 x2 = ((const float4*)(e2 + r2))[lane];
      s += a1.x * x2.x + a1.y * x2.y + a1.z * x2.z + a1.w * x2.w;
    }
    total += s / (float)(hi - lo);
  }

#pragma unroll
  for (int m = 16; m >= 1; m >>= 1) total += __shfl_xor(total, m, 32);
  if (lane == 0) pos[row] = total;
}

// ---------------------------------------------------------------------------
// Fused sim GEMM + streaming logsumexp, partial over a column chunk.
//   Wave owns 32 rows (2 row-tiles -> 8 resident A-frags, 8 WMMAs per B-tile).
//   gridDim.y = CSPLIT column chunks; each wave streams N/CSPLIT columns.
//   Online softmax state is PER-LANE (lane ln sees cols ≡ ln mod 16): no
//   cross-lane ops in the hot loop; one width-16 (max,sum) merge at the end.
//   Emits per-(row,chunk) partial (max2, Σ2^(x-max2)) in base-2 domain.
// block = 256 (8 waves) ; grid = (N/256, CSPLIT)
// ---------------------------------------------------------------------------
__global__ void sim_lse_kernel(const unsigned short* __restrict__ e1b,
                               const unsigned short* __restrict__ e2b,
                               float* __restrict__ pmax,
                               float* __restrict__ psum, int N) {
  const int lane = threadIdx.x & 31;
  const int wave = threadIdx.x >> 5;
  const int h = lane >> 4;
  const int ln = lane & 15;
  const int rbase = (blockIdx.x * 8 + wave) * 32;   // 32 rows per wave
  const int chunk = blockIdx.y;
  const int cols = N / CSPLIT;                      // 1024
  const int cbase = chunk * cols;
  const int DF = 128;

  // resident A fragments: 2 row-tiles x 4 K-chunks
  v16bf afr[2][4];
#pragma unroll
  for (int t = 0; t < 2; ++t)
#pragma unroll
    for (int kc = 0; kc < 4; ++kc)
      afr[t][kc] = load_a_frag(e1b, rbase + t * 16 + ln, DF, kc * 32, h);

  // per-lane online softmax state (base-2): slot (t,r) covers row
  // rbase + t*16 + r + 8h, columns ≡ ln (mod 16) within this chunk
  float m[2][8], s[2][8];
#pragma unroll
  for (int t = 0; t < 2; ++t)
#pragma unroll
    for (int r = 0; r < 8; ++r) { m[t][r] = -__builtin_inff(); s[t][r] = 0.f; }

  const int ntiles = cols / 16;                     // 64
  for (int ct = 0; ct < ntiles; ++ct) {
    const int col = cbase + ct * 16 + ln;
    if (ct + 1 < ntiles)
      __builtin_prefetch((const void*)(e2b + (size_t)(col + 16) * DF), 0, 1);

    v8f acc0 = {}, acc1 = {};
#pragma unroll
    for (int kc = 0; kc < 4; ++kc) {
      v16bf b = load_b_frag(e2b, col, DF, kc * 32, h);
      acc0 = __builtin_amdgcn_wmma_f32_16x16x32_bf16(false, afr[0][kc], false, b,
                                                     (short)0, acc0, false, false);
      acc1 = __builtin_amdgcn_wmma_f32_16x16x32_bf16(false, afr[1][kc], false, b,
                                                     (short)0, acc1, false, false);
    }

    // per-lane online update; common path = 1 exp2 + compare per element
#pragma unroll
    for (int t = 0; t < 2; ++t) {
#pragma unroll
      for (int r = 0; r < 8; ++r) {
        float v = (t == 0 ? acc0[r] : acc1[r]) * LOG2E_F;
        if (v <= m[t][r]) {
          s[t][r] += exp2f(v - m[t][r]);
        } else {
          s[t][r] = s[t][r] * exp2f(m[t][r] - v) + 1.0f;
          m[t][r] = v;
        }
      }
    }
  }

  // merge the 16 per-lane states of each half-wave (once per wave)
#pragma unroll
  for (int t = 0; t < 2; ++t) {
#pragma unroll
    for (int r = 0; r < 8; ++r) {
      float mm = m[t][r], ss = s[t][r];
#pragma unroll
      for (int d = 1; d <= 8; d <<= 1) {
        float om = __shfl_xor(mm, d, 16);
        float os = __shfl_xor(ss, d, 16);
        float nm = fmaxf(mm, om);
        ss = ss * exp2f(mm - nm) + os * exp2f(om - nm);
        mm = nm;
      }
      if (ln == 0) {
        int row = rbase + t * 16 + r + 8 * h;
        size_t idx = (size_t)row * CSPLIT + chunk;
        pmax[idx] = mm;
        psum[idx] = ss;
      }
    }
  }
}

// ---------------------------------------------------------------------------
// Final reduction: merge per-chunk (max,sum) partials exactly, then
//   loss = mean_i( lse_i - log(N) - positive[i] )
// One thread per row (grid*block == N).
// ---------------------------------------------------------------------------
__global__ void zero_kernel(float* out) { out[0] = 0.f; }

__global__ void finalize_kernel(const float* __restrict__ pos,
                                const float* __restrict__ pmax,
                                const float* __restrict__ psum,
                                float* __restrict__ out, int N, float logN) {
  __shared__ float sbuf[8];
  const int i = blockIdx.x * blockDim.x + threadIdx.x;

  float v = 0.f;
  if (i < N) {
    float M = -__builtin_inff();
#pragma unroll
    for (int c = 0; c < CSPLIT; ++c) M = fmaxf(M, pmax[(size_t)i * CSPLIT + c]);
    float S = 0.f;
#pragma unroll
    for (int c = 0; c < CSPLIT; ++c)
      S += psum[(size_t)i * CSPLIT + c] * exp2f(pmax[(size_t)i * CSPLIT + c] - M);
    float lse = (M + log2f(S)) * LN2_F;
    v = lse - logN - pos[i];
  }

#pragma unroll
  for (int mk = 16; mk >= 1; mk >>= 1) v += __shfl_xor(v, mk, 32);
  const int lane = threadIdx.x & 31, wave = threadIdx.x >> 5;
  if (lane == 0) sbuf[wave] = v;
  __syncthreads();
  if (wave == 0) {
    v = (lane < 8) ? sbuf[lane] : 0.f;
#pragma unroll
    for (int mk = 4; mk >= 1; mk >>= 1) v += __shfl_xor(v, mk, 32);
    if (lane == 0) atomicAdd(out, v / (float)N);
  }
}

// ---------------------------------------------------------------------------
extern "C" void kernel_launch(void* const* d_in, const int* in_sizes, int n_in,
                              void* d_out, int out_size, void* d_ws, size_t ws_size,
                              hipStream_t stream) {
  const int B = 16, L = 512, DIN1 = 256, DIN2 = 192, DH = 256, DF = 128;
  const int N = B * L;  // 8192

  const float* f1  = (const float*)d_in[0];
  const float* f2  = (const float*)d_in[1];
  const float* W1a = (const float*)d_in[2];
  const float* b1a = (const float*)d_in[3];
  const float* W2a = (const float*)d_in[4];
  const float* b2a = (const float*)d_in[5];
  const float* W1b = (const float*)d_in[6];
  const float* b1b = (const float*)d_in[7];
  const float* W2b = (const float*)d_in[8];
  const float* b2b = (const float*)d_in[9];
  const int* prs = (const int*)d_in[10];
  const int* prt = (const int*)d_in[11];
  float* out = (float*)d_out;
  (void)in_sizes; (void)n_in; (void)out_size; (void)ws_size;

  char* ws = (char*)d_ws;
  size_t off = 0;
  auto alloc = [&](size_t bytes) -> char* {
    char* p = ws + off;
    off += (bytes + 255) & ~(size_t)255;
    return p;
  };
  unsigned short* f1b  = (unsigned short*)alloc((size_t)N * DIN1 * 2);
  unsigned short* f2b  = (unsigned short*)alloc((size_t)N * DIN2 * 2);
  unsigned short* W1at = (unsigned short*)alloc((size_t)DH * DIN1 * 2);
  unsigned short* W2at = (unsigned short*)alloc((size_t)DF * DH * 2);
  unsigned short* W1bt = (unsigned short*)alloc((size_t)DH * DIN2 * 2);
  unsigned short* W2bt = (unsigned short*)alloc((size_t)DF * DH * 2);
  unsigned short* Hbuf = (unsigned short*)alloc((size_t)N * DH * 2);
  float*          e1f  = (float*)alloc((size_t)N * DF * 4);
  float*          e2f  = (float*)alloc((size_t)N * DF * 4);
  unsigned short* e1b  = (unsigned short*)alloc((size_t)N * DF * 2);
  unsigned short* e2b  = (unsigned short*)alloc((size_t)N * DF * 2);
  float*          pos  = (float*)alloc((size_t)N * 4);
  float*          pmax = (float*)alloc((size_t)N * CSPLIT * 4);
  float*          psum = (float*)alloc((size_t)N * CSPLIT * 4);

  // --- convert inputs / transpose weights to bf16 ---
  cvt_bf16_kernel<<<(N * DIN1 + 255) / 256, 256, 0, stream>>>(f1, f1b, N * DIN1);
  cvt_bf16_kernel<<<(N * DIN2 + 255) / 256, 256, 0, stream>>>(f2, f2b, N * DIN2);
  transpose_bf16_kernel<<<(DIN1 * DH + 255) / 256, 256, 0, stream>>>(W1a, W1at, DIN1, DH);
  transpose_bf16_kernel<<<(DH * DF + 255) / 256, 256, 0, stream>>>(W2a, W2at, DH, DF);
  transpose_bf16_kernel<<<(DIN2 * DH + 255) / 256, 256, 0, stream>>>(W1b, W1bt, DIN2, DH);
  transpose_bf16_kernel<<<(DH * DF + 255) / 256, 256, 0, stream>>>(W2b, W2bt, DH, DF);

  // --- MLP 1: H = relu(f1@W1a+b1a); e1 = H@W2a+b2a ---
  gemm_bf16_kernel<<<dim3(DH / 64, N / 16), 128, 0, stream>>>(
      f1b, W1at, b1a, nullptr, Hbuf, DIN1, DH, 1);
  gemm_bf16_kernel<<<dim3(DF / 64, N / 16), 128, 0, stream>>>(
      Hbuf, W2at, b2a, e1f, e1b, DH, DF, 0);
  // --- MLP 2 (reuse Hbuf) ---
  gemm_bf16_kernel<<<dim3(DH / 64, N / 16), 128, 0, stream>>>(
      f2b, W1bt, b1b, nullptr, Hbuf, DIN2, DH, 1);
  gemm_bf16_kernel<<<dim3(DF / 64, N / 16), 128, 0, stream>>>(
      Hbuf, W2bt, b2b, e2f, e2b, DH, DF, 0);

  // --- positive + band terms (exact f32) ---
  band_positive_kernel<<<N / 8, 256, 0, stream>>>(e1f, e2f, pos, prs, prt, L);

  // --- fused N x N sim + streaming logsumexp partials (dominant op) ---
  sim_lse_kernel<<<dim3(N / 256, CSPLIT), 256, 0, stream>>>(e1b, e2b, pmax, psum, N);

  // --- final merge + mean reduction ---
  zero_kernel<<<1, 1, 0, stream>>>(out);
  finalize_kernel<<<N / 256, 256, 0, stream>>>(pos, pmax, psum, out, N, logf((float)N));
}
